// DeepGpcmModel_65283502899476
// MI455X (gfx1250) — compile-verified, matrix operand order checked by reading, not run on hardware
//
#include <hip/hip_runtime.h>
#include <hip/hip_bf16.h>

#define BATCH 1024
#define SEQ   500
#define NCAT  4
#define MEM   50
#define KD    50
#define VD    200
#define FC    50
#define NQ    2000

// K-pads chosen so (kpad/2) is ODD -> odd dword stride between rows -> LDS
// bank-conflict-free fragment gathers (banks = (addr/4) % 64).
#define KD_P   66    // qe K-pad (attention GEMM K covers 64), Mk kpad
#define MEM_P  64    // attention N-pad
#define VE_KP  226   // ve K-pad (erase/add GEMM K covers 224)
#define VE_NP  208   // erase/add N-pad (13 tiles)
#define SUM_KP 258   // concat(read,qe) K-pad (K covers 256)
#define FC_NP  64    // summary N-pad

// f16 element counts of the staged weight images (contiguous in ws and in LDS)
#define SZ_WE  (VE_NP * VE_KP)   // 47008 elems, 94016 B
#define SZ_WS  (FC_NP * SUM_KP)  // 16512 elems, 33024 B
#define SZ_MK  (MEM_P * KD_P)    //  4224 elems,  8448 B
#define WSTAGE_BYTES ((size_t)SZ_WE * 4 + (size_t)SZ_WS * 2 + (size_t)SZ_MK * 2) // 229504
#define WSTAGE_V16   (WSTAGE_BYTES / 16)  // 14344 b128 chunks
// byte offsets inside the staged region
#define OFF_WE  0
#define OFF_WA  (SZ_WE * 2)
#define OFF_WS  (SZ_WE * 4)
#define OFF_MK  (SZ_WE * 4 + SZ_WS * 2)

typedef __attribute__((ext_vector_type(16))) _Float16 v16h;
typedef __attribute__((ext_vector_type(8)))  float    v8f;

// Load one 16x32 f16 fragment (A layout; same layout used for B via B^T storage).
// src: row-major [rows][kpad] f16, 4-byte aligned rows, kpad even.
__device__ __forceinline__ v16h load_frag(const _Float16* src, int kpad, int lane, int kb0) {
  union { v16h h; unsigned int u[8]; } f;
  const int half = (lane >> 4) & 1;
  const int r    = lane & 15;
  const unsigned int* p = (const unsigned int*)(src + (size_t)r * kpad);
  const int d0 = (kb0 + half * 8) >> 1;        // dword index of first pair
  const int d1 = (kb0 + 16 + half * 8) >> 1;
#pragma unroll
  for (int i = 0; i < 4; ++i) { f.u[i] = p[d0 + i]; f.u[4 + i] = p[d1 + i]; }
  return f.h;
}

__device__ __forceinline__ v8f wmma_f16(v16h a, v16h b, v8f c) {
  return __builtin_amdgcn_wmma_f32_16x16x32_f16(false, a, false, b, (short)0, c, false, false);
}

// CDNA5 async global->LDS copy (VGLOBAL GLOBAL_LOAD_ASYNC_TO_LDS_B128, ASYNCcnt).
__device__ __forceinline__ void async_g2l_b128(unsigned lds_addr, const void* gaddr) {
  asm volatile("global_load_async_to_lds_b128 %0, %1, off"
               :: "v"(lds_addr), "v"((unsigned long long)gaddr) : "memory");
}
__device__ __forceinline__ void wait_asynccnt0() {
  asm volatile("s_wait_asynccnt 0" ::: "memory");
}

// ---- prep kernels -----------------------------------------------------------

__global__ void convert_pad_f16(const float* __restrict__ src, _Float16* __restrict__ dst,
                                int rows, int cols, int cpad, int total) {
  int i = blockIdx.x * 256 + threadIdx.x;
  if (i >= total) return;
  int r = i / cpad, c = i - r * cpad;
  float v = (r < rows && c < cols) ? src[(size_t)r * cols + c] : 0.0f;
  dst[i] = (_Float16)v;
}

__global__ void init_mv_kernel(const float* __restrict__ mv0, float* __restrict__ mv, int total) {
  int i = blockIdx.x * 256 + threadIdx.x;
  if (i < total) mv[i] = mv0[i % (MEM * VD)];
}

// ---- main persistent scan kernel: one block = 16 batch rows, all 500 steps --
// Dynamic LDS holds ALL f16 weight matrices for the whole run (WGP has 320KB).

extern __shared__ __align__(16) char s_dyn[];

__global__ __launch_bounds__(256)
void deep_gpcm_main(const int* __restrict__ qd, const int* __restrict__ rd,
                    const float* __restrict__ qemb,
                    const float* __restrict__ p_bv, const float* __restrict__ Wv,
                    const float* __restrict__ p_be, const float* __restrict__ p_ba,
                    const float* __restrict__ p_bs,
                    const float* __restrict__ Wth,  const float* __restrict__ p_bth,
                    const float* __restrict__ Wbeta,const float* __restrict__ p_bbeta,
                    const float* __restrict__ Wdisc,const float* __restrict__ p_bdisc,
                    const char* __restrict__ wstage,
                    float* __restrict__ Mv, float* __restrict__ out) {
  __shared__ float s_qe[16 * 52];
  __shared__ float s_cw[16 * 64];
  __shared__ float s_sum[16 * 64];
  __shared__ float s_w[16 * 4];
  __shared__ int   s_qi[16], s_ri[16];
  __shared__ __align__(16) _Float16 s_Aq[16 * KD_P];
  __shared__ __align__(16) _Float16 s_Av[16 * VE_KP];
  __shared__ __align__(16) _Float16 s_As[16 * SUM_KP];

  const int tid  = threadIdx.x;
  const int wave = tid >> 5;
  const int lane = tid & 31;
  const int bg0  = blockIdx.x * 16;

  // ---- one-time async stage of all weight matrices into LDS ----
  {
    const unsigned lds_base = (unsigned)(size_t)(void*)s_dyn;  // low 32 bits = LDS offset
    for (int idx = tid; idx < (int)WSTAGE_V16; idx += 256)
      async_g2l_b128(lds_base + (unsigned)idx * 16u, wstage + (size_t)idx * 16u);
    wait_asynccnt0();
    __syncthreads();
  }
  const _Float16* lWe = (const _Float16*)(s_dyn + OFF_WE);
  const _Float16* lWa = (const _Float16*)(s_dyn + OFF_WA);
  const _Float16* lWs = (const _Float16*)(s_dyn + OFF_WS);
  const _Float16* lMk = (const _Float16*)(s_dyn + OFF_MK);

  for (int t = 0; t < SEQ; ++t) {
    // ---- phase A: fetch q/r, question embedding, decay weights, value embedding
    if (tid < 16) {
      s_qi[tid] = qd[(size_t)(bg0 + tid) * SEQ + t];
      s_ri[tid] = rd[(size_t)(bg0 + tid) * SEQ + t];
    }
    __syncthreads();
    for (int idx = tid; idx < 16 * KD_P; idx += 256) {
      int b = idx / KD_P, d = idx - b * KD_P;
      _Float16 hv = (_Float16)0.0f;
      if (d < KD) {
        float v = qemb[(size_t)s_qi[b] * KD + d];
        s_qe[b * 52 + d] = v;
        hv = (_Float16)v;
      }
      s_Aq[idx] = hv;
    }
    if (tid < 64) {
      int b = tid >> 2, k = tid & 3;
      float wv = 1.0f - fabsf((float)k - (float)s_ri[b]) * (1.0f / 3.0f);
      s_w[tid] = fmaxf(0.0f, wv);
    }
    __syncthreads();
    for (int idx = tid; idx < 16 * VE_KP; idx += 256) {
      int b = idx / VE_KP, d = idx - b * VE_KP;
      _Float16 hv = (_Float16)0.0f;
      if (d < VD) {
        int q = s_qi[b] - 1;
        float v = p_bv[d];
#pragma unroll
        for (int k = 0; k < NCAT; ++k)
          v += s_w[b * 4 + k] * Wv[(size_t)d * (NCAT * NQ) + k * NQ + q];
        hv = (_Float16)v;
      }
      s_Av[idx] = hv;
    }
    __syncthreads();

    // ---- phase B: attention logits = qe @ Mk^T  (WMMA, waves 0..3, B from LDS)
    if (wave < 4) {
      const int nb = wave * 16;
      v8f acc = {};
#pragma unroll
      for (int kt = 0; kt < 2; ++kt) {
        v16h a  = load_frag(s_Aq, KD_P, lane, kt * 32);
        v16h bm = load_frag(lMk + (size_t)nb * KD_P, KD_P, lane, kt * 32);
        acc = wmma_f16(a, bm, acc);
      }
      union { v8f v; float f[8]; } u; u.v = acc;
      const int n = nb + (lane & 15);
      const int hh = (lane >> 4) & 1;
      if (n < MEM) {
#pragma unroll
        for (int j = 0; j < 8; ++j) s_cw[(j + 8 * hh) * 64 + n] = u.f[j];
      }
    }
    __syncthreads();

    // ---- softmax over memory slots (one row per lane of wave 0)
    if (tid < 16) {
      float mx = -1e30f;
      for (int i = 0; i < MEM; ++i) mx = fmaxf(mx, s_cw[tid * 64 + i]);
      float sum = 0.0f;
      for (int i = 0; i < MEM; ++i) { float e = __expf(s_cw[tid * 64 + i] - mx); s_cw[tid * 64 + i] = e; sum += e; }
      float inv = 1.0f / sum;
      for (int i = 0; i < MEM; ++i) s_cw[tid * 64 + i] *= inv;
    }
    __syncthreads();

    // ---- phase C: memory read (batched, VALU vs L2-resident Mv) + concat A for summary
    for (int idx = tid; idx < 16 * SUM_KP; idx += 256) {
      int b = idx / SUM_KP, c = idx - b * SUM_KP;
      _Float16 hv = (_Float16)0.0f;
      if (c < VD) {
        const float* mvb = Mv + (size_t)(bg0 + b) * (MEM * VD) + c;
        float r = 0.0f;
        for (int m = 0; m < MEM; ++m) r += s_cw[b * 64 + m] * mvb[(size_t)m * VD];
        hv = (_Float16)r;
      } else if (c < VD + KD) {
        hv = (_Float16)s_qe[b * 52 + (c - VD)];
      }
      s_As[idx] = hv;
    }
    __syncthreads();

    // ---- phase D: summary = tanh(concat @ Ws^T + bs)  (WMMA, waves 0..3, B from LDS)
    if (wave < 4) {
      const int nb = wave * 16;
      v8f acc = {};
#pragma unroll
      for (int kt = 0; kt < 8; ++kt) {
        v16h a  = load_frag(s_As, SUM_KP, lane, kt * 32);
        v16h bm = load_frag(lWs + (size_t)nb * SUM_KP, SUM_KP, lane, kt * 32);
        acc = wmma_f16(a, bm, acc);
      }
      union { v8f v; float f[8]; } u; u.v = acc;
      const int n = nb + (lane & 15);
      const int hh = (lane >> 4) & 1;
      if (n < FC) {
        float bsv = p_bs[n];
#pragma unroll
        for (int j = 0; j < 8; ++j) s_sum[(j + 8 * hh) * 64 + n] = tanhf(u.f[j] + bsv);
      }
    }
    __syncthreads();

    // ---- phase E: IRT heads + GPCM probs -> output (tiny, lane-per-row)
    if (tid < 16) {
      const int b = tid;
      float th = p_bth[0];
      for (int i = 0; i < FC; ++i) th += s_sum[b * 64 + i] * Wth[i];
      th *= 3.0f;
      float al = p_bdisc[0];
      for (int i = 0; i < FC; ++i) al += s_sum[b * 64 + i] * Wdisc[i];
      for (int i = 0; i < KD; ++i) al += s_qe[b * 52 + i] * Wdisc[FC + i];
      al = (al > 20.0f) ? al : log1pf(__expf(al));
      float bet[3];
      for (int k = 0; k < NCAT - 1; ++k) {
        float v = p_bbeta[k];
        for (int i = 0; i < KD; ++i) v += s_qe[b * 52 + i] * Wbeta[k * KD + i];
        bet[k] = tanhf(v);
      }
      float lg[4]; lg[0] = 0.0f;
      float c = 0.0f;
      for (int k = 0; k < 3; ++k) { c += al * (th - bet[k]); lg[k + 1] = c; }
      float mx = lg[0];
      for (int k = 1; k < 4; ++k) mx = fmaxf(mx, lg[k]);
      float e[4], sum = 0.0f;
      for (int k = 0; k < 4; ++k) { e[k] = __expf(lg[k] - mx); sum += e[k]; }
      float inv = 1.0f / sum;
      float* op = out + ((size_t)(bg0 + b) * SEQ + t) * NCAT;
      for (int k = 0; k < 4; ++k) op[k] = e[k] * inv;
    }

    // ---- phase F: erase/add GEMMs (B from LDS) + fused erase-add memory write
    for (int nt = wave; nt < 13; nt += 8) {
      const int nb = nt * 16;
      v8f ae = {}, aa = {};
#pragma unroll
      for (int kt = 0; kt < 7; ++kt) {
        v16h a  = load_frag(s_Av, VE_KP, lane, kt * 32);
        v16h bE = load_frag(lWe + (size_t)nb * VE_KP, VE_KP, lane, kt * 32);
        ae = wmma_f16(a, bE, ae);
        v16h bA = load_frag(lWa + (size_t)nb * VE_KP, VE_KP, lane, kt * 32);
        aa = wmma_f16(a, bA, aa);
      }
      union { v8f v; float f[8]; } ue, ua; ue.v = ae; ua.v = aa;
      const int n  = nb + (lane & 15);
      const int hh = (lane >> 4) & 1;
      if (n < VD) {
        const float bev = p_be[n], bav = p_ba[n];
#pragma unroll
        for (int j = 0; j < 8; ++j) {
          const int b = j + 8 * hh;
          float er = 1.0f / (1.0f + __expf(-(ue.f[j] + bev)));
          float ad = tanhf(ua.f[j] + bav);
          float* mv = Mv + (size_t)(bg0 + b) * (MEM * VD) + n;
          for (int m = 0; m < MEM; ++m) {
            float cv  = s_cw[b * 64 + m];
            float old = mv[(size_t)m * VD];
            mv[(size_t)m * VD] = old * (1.0f - cv * er) + cv * ad;
          }
        }
      }
    }
    __threadfence();   // Mv writes visible for next step's phase C
    __syncthreads();
  }
}

// ---- host-side launch -------------------------------------------------------

extern "C" void kernel_launch(void* const* d_in, const int* in_sizes, int n_in,
                              void* d_out, int out_size, void* d_ws, size_t ws_size,
                              hipStream_t stream) {
  const int*   qd    = (const int*)d_in[0];
  const int*   rd    = (const int*)d_in[1];
  const float* qemb  = (const float*)d_in[2];
  const float* Mk    = (const float*)d_in[3];
  const float* Mv0   = (const float*)d_in[4];
  const float* Wv    = (const float*)d_in[5];
  const float* bv    = (const float*)d_in[6];
  const float* We    = (const float*)d_in[7];
  const float* be    = (const float*)d_in[8];
  const float* Wa    = (const float*)d_in[9];
  const float* ba    = (const float*)d_in[10];
  const float* Ws    = (const float*)d_in[11];
  const float* bs    = (const float*)d_in[12];
  const float* Wth   = (const float*)d_in[13];
  const float* bth   = (const float*)d_in[14];
  const float* Wbeta = (const float*)d_in[15];
  const float* bbeta = (const float*)d_in[16];
  const float* Wdisc = (const float*)d_in[17];
  const float* bdisc = (const float*)d_in[18];
  float* out = (float*)d_out;

  // workspace carve-up: contiguous f16 weight staging image + persistent Mv state
  char* ws = (char*)d_ws;
  _Float16* Wef = (_Float16*)(ws + OFF_WE);
  _Float16* Waf = (_Float16*)(ws + OFF_WA);
  _Float16* Wsf = (_Float16*)(ws + OFF_WS);
  _Float16* Mkf = (_Float16*)(ws + OFF_MK);
  size_t off = (WSTAGE_BYTES + 255) & ~(size_t)255;
  float* Mv = (float*)(ws + off);                       // 40.96 MB, L2-resident

  convert_pad_f16<<<(SZ_WE + 255) / 256, 256, 0, stream>>>(We, Wef, VD, VD, VE_KP, SZ_WE);
  convert_pad_f16<<<(SZ_WE + 255) / 256, 256, 0, stream>>>(Wa, Waf, VD, VD, VE_KP, SZ_WE);
  convert_pad_f16<<<(SZ_WS + 255) / 256, 256, 0, stream>>>(Ws, Wsf, FC, VD + KD, SUM_KP, SZ_WS);
  convert_pad_f16<<<(SZ_MK + 255) / 256, 256, 0, stream>>>(Mk, Mkf, MEM, KD, KD_P, SZ_MK);
  const int tMv = BATCH * MEM * VD;
  init_mv_kernel<<<(tMv + 255) / 256, 256, 0, stream>>>(Mv0, Mv, tMv);

  deep_gpcm_main<<<BATCH / 16, 256, (unsigned)WSTAGE_BYTES, stream>>>(
      qd, rd, qemb, bv, Wv, be, ba, bs, Wth, bth, Wbeta, bbeta, Wdisc, bdisc,
      (const char*)ws, Mv, out);
}
